// GraphFusion_36146444763302
// MI455X (gfx1250) — matrix operand order repaired and checked by reading, not compiled, not run on hardware
//
#include <hip/hip_runtime.h>
#include <hip/hip_bf16.h>
#include <math.h>

// ---------------------------------------------------------------------------
// Problem sizes (fixed by the reference)
// ---------------------------------------------------------------------------
#define NM 8192   // num_node
#define SS 256    // num_state
#define HH 128    // query/key channels
#define NTHREADS 256

typedef __attribute__((ext_vector_type(16))) __bf16 bf16x16;
typedef __attribute__((ext_vector_type(8)))  __bf16 bf16x8;
typedef __attribute__((ext_vector_type(2)))  __bf16 bf16x2;
typedef __attribute__((ext_vector_type(8)))  float  f32x8;

// ---------------------------------------------------------------------------
// WMMA GEMM: C[M,N] = act( op(A)[M,K] * op(B)[K,N] + bias + residual )
//   - A,B,C,resid fp32 in HBM; converted to bf16 while staging into LDS, so
//     the math is v_wmma_f32_16x16x32_bf16 with f32 accumulation.
//   - LDS tiles are stored in per-lane fragment order, k contiguous in each lane half:
//       A sub-tile 16x32:  As[tile][lane][i]  holds A[m=lane&15]
//                          [k=(lane>>4)*16+i]
//       B sub-tile 32x16:  Bs[tile][lane][i]  holds B[k=(lane>>4)*16+i]
//                          [n=lane&15]
//     so each fragment load is ONE contiguous 32B bf16x16 LDS read.
//   - Scatter (transpose-ish) staging paths pair adjacent k rows so every
//     store is a packed-bf16 ds_store_b32 (v_cvt_pk_bf16_f32 + b32 store).
//   - Double-buffered LDS; next tile prefetched into VGPRs before the WMMAs.
//   - AT: logical A[m,k]=Asrc[k*lda+m];  BT: logical B[k,n]=Bsrc[n*ldb+k]
//   - bias_mode: 0 none, 1 per-row bias[m], 2 per-col bias[n]
//   - All M,N multiples of 128, K multiples of 32 here -> no edge guards.
// Block tile 128x128x32, 256 threads = 8 waves, each wave owns 32x64
// (2x4 tiles of 16x16).
// ---------------------------------------------------------------------------
#define BM 128
#define BN 128
#define BK 32
#define TS 528   // padded sub-tile stride (bf16 elems); 528*2B = 32B-aligned

template <bool AT, bool BT>
__global__ __launch_bounds__(NTHREADS)
void gemm_wmma(const float* __restrict__ A, int lda,
               const float* __restrict__ B, int ldb,
               float* __restrict__ C, int ldc,
               const float* __restrict__ bias, int bias_mode,
               const float* __restrict__ resid, int do_relu,
               int M, int N, int K)
{
    __shared__ __bf16 As[2][8 * TS];
    __shared__ __bf16 Bs[2][8 * TS];

    const int tid  = threadIdx.x;
    const int wave = tid >> 5;
    const int lane = tid & 31;
    const int bm   = blockIdx.y * BM;
    const int bn   = blockIdx.x * BN;
    const int mt0  = (wave & 3) * 2;    // first of this wave's 2 m sub-tiles
    const int nt0  = (wave >> 2) * 4;   // first of this wave's 4 n sub-tiles

    // thread->data mapping for the paired-k scatter paths
    const int kp  = (tid >> 4) << 1;    // even k row: 0,2,...,30
    const int c8  = (tid & 15) << 3;    // 8-wide column group: 0,8,...,120

    f32x8 acc[2][4];
#pragma unroll
    for (int mi = 0; mi < 2; ++mi)
#pragma unroll
        for (int ni = 0; ni < 4; ++ni)
            acc[mi][ni] = f32x8{};

    float4 fa[4], fb[4];

    // ---- global tile -> VGPR prefetch ----
    auto g_load = [&](int k0) {
        if (!AT) {
            // row m = tid>>1, 16 consecutive k
            const float* s = A + (size_t)(bm + (tid >> 1)) * lda + k0 + (tid & 1) * 16;
#pragma unroll
            for (int v = 0; v < 4; ++v) fa[v] = *(const float4*)(s + 4 * v);
        } else {
            // rows k = kp, kp+1; 8 consecutive m at bm+c8
            const float* s = A + (size_t)(k0 + kp) * lda + bm + c8;
            fa[0] = *(const float4*)(s + 0);
            fa[1] = *(const float4*)(s + 4);
            fa[2] = *(const float4*)(s + lda + 0);
            fa[3] = *(const float4*)(s + lda + 4);
        }
        if (!BT) {
            // rows k = kp, kp+1; 8 consecutive n at bn+c8
            const float* s = B + (size_t)(k0 + kp) * ldb + bn + c8;
            fb[0] = *(const float4*)(s + 0);
            fb[1] = *(const float4*)(s + 4);
            fb[2] = *(const float4*)(s + ldb + 0);
            fb[3] = *(const float4*)(s + ldb + 4);
        } else {
            // row n = tid>>1, 16 consecutive k
            const float* s = B + (size_t)(bn + (tid >> 1)) * ldb + k0 + (tid & 1) * 16;
#pragma unroll
            for (int v = 0; v < 4; ++v) fb[v] = *(const float4*)(s + 4 * v);
        }
    };

    // ---- convert VGPR tile -> fragment-ordered bf16 LDS ----
    auto stage = [&](int buf) {
        if (!AT) {
            // thread holds A[m = tid>>1][k16..k16+15]: two contiguous 8-runs
            __bf16 ha[16];
#pragma unroll
            for (int v = 0; v < 4; ++v) {
                ha[4 * v + 0] = (__bf16)fa[v].x;  ha[4 * v + 1] = (__bf16)fa[v].y;
                ha[4 * v + 2] = (__bf16)fa[v].z;  ha[4 * v + 3] = (__bf16)fa[v].w;
            }
            const int m = tid >> 1, g8 = ((tid & 1) << 8);
            __bf16* d0 = &As[buf][(m >> 4) * TS + ((m & 15) << 4) + g8];
            __bf16* d1 = d0 + 8;
            bf16x8 lo, hi;
#pragma unroll
            for (int j = 0; j < 8; ++j) { lo[j] = ha[j]; hi[j] = ha[8 + j]; }
            *(bf16x8*)d0 = lo;
            *(bf16x8*)d1 = hi;
        } else {
            // thread holds A[kp..kp+1][m = c8..c8+7]: per m, elems (eA,eA+1)
            // are adjacent -> packed b32 stores
            const int halfA = kp >> 4;
            const int eA    = kp & 15;                        // even
            const float* r0 = (const float*)&fa[0];          // row kp
            const float* r1 = (const float*)&fa[2];          // row kp+1
            __bf16* base = &As[buf][(c8 >> 4) * TS + (halfA << 8) + eA];
#pragma unroll
            for (int j = 0; j < 8; ++j) {
                bf16x2 p; p[0] = (__bf16)r0[j]; p[1] = (__bf16)r1[j];
                *(bf16x2*)&base[((c8 + j) & 15) << 4] = p;
            }
        }
        if (!BT) {
            // thread holds B[kp..kp+1][n = c8..c8+7]: packed b32 stores
            const int halfB = kp >> 4;
            const int eB    = kp & 15;                        // even
            const float* r0 = (const float*)&fb[0];
            const float* r1 = (const float*)&fb[2];
            __bf16* base = &Bs[buf][(c8 >> 4) * TS + (halfB << 8) + eB];
#pragma unroll
            for (int j = 0; j < 8; ++j) {
                bf16x2 p; p[0] = (__bf16)r0[j]; p[1] = (__bf16)r1[j];
                *(bf16x2*)&base[((c8 + j) & 15) << 4] = p;
            }
        } else {
            // thread holds B[k = kk0..kk0+15][n = tid>>1]: fully contiguous
            __bf16 hb[16];
#pragma unroll
            for (int v = 0; v < 4; ++v) {
                hb[4 * v + 0] = (__bf16)fb[v].x;  hb[4 * v + 1] = (__bf16)fb[v].y;
                hb[4 * v + 2] = (__bf16)fb[v].z;  hb[4 * v + 3] = (__bf16)fb[v].w;
            }
            const int n = tid >> 1, halfB = tid & 1;
            __bf16* d = &Bs[buf][(n >> 4) * TS + ((halfB << 8) + ((n & 15) << 4))];
            bf16x8 lo, hi;
#pragma unroll
            for (int j = 0; j < 8; ++j) { lo[j] = hb[j]; hi[j] = hb[8 + j]; }
            *(bf16x8*)(d + 0) = lo;
            *(bf16x8*)(d + 8) = hi;
        }
    };

    // ---- one k-step of WMMAs from fragment-ordered LDS ----
    auto compute = [&](int buf) {
        bf16x16 af[2], bf[4];
#pragma unroll
        for (int mi = 0; mi < 2; ++mi)
            af[mi] = *(const bf16x16*)&As[buf][(mt0 + mi) * TS + (lane << 4)];
#pragma unroll
        for (int ni = 0; ni < 4; ++ni)
            bf[ni] = *(const bf16x16*)&Bs[buf][(nt0 + ni) * TS + (lane << 4)];
#pragma unroll
        for (int mi = 0; mi < 2; ++mi)
#pragma unroll
            for (int ni = 0; ni < 4; ++ni)
                acc[mi][ni] = __builtin_amdgcn_wmma_f32_16x16x32_bf16(
                    false, af[mi], false, bf[ni],
                    (short)0, acc[mi][ni], false, false);
    };

    const int nk = K / BK;
    g_load(0);
    stage(0);
    for (int kt = 0; kt < nk; ++kt) {
        const int buf = kt & 1;
        __syncthreads();                         // LDS[buf] ready / buf^1 free
        if (kt + 1 < nk) g_load((kt + 1) * BK);  // prefetch next tile
        compute(buf);                            // 8 WMMAs cover load latency
        if (kt + 1 < nk) stage(buf ^ 1);
    }

    // ---- epilogue: bias / residual / relu per C-frag layout (ISA 7.12.2) ----
    const int ncol = lane & 15, half = lane >> 4;
#pragma unroll
    for (int mi = 0; mi < 2; ++mi) {
#pragma unroll
        for (int ni = 0; ni < 4; ++ni) {
#pragma unroll
            for (int v = 0; v < 8; ++v) {
                const int row = bm + (mt0 + mi) * 16 + v + 8 * half;
                const int col = bn + (nt0 + ni) * 16 + ncol;
                float x = acc[mi][ni][v];
                if (bias_mode == 1)      x += bias[row];
                else if (bias_mode == 2) x += bias[col];
                if (resid)               x += resid[(size_t)row * ldc + col];
                if (do_relu)             x = fmaxf(x, 0.0f);
                C[(size_t)row * ldc + col] = x;
            }
        }
    }
}

// ---------------------------------------------------------------------------
// Row softmax (in-place capable): O[r,:] = softmax(E[r,:]); one block per row.
// ---------------------------------------------------------------------------
__global__ __launch_bounds__(NTHREADS)
void row_softmax(const float* __restrict__ E, float* __restrict__ O, int N)
{
    const int row = blockIdx.x;
    const int tid = threadIdx.x;
    const float* e = E + (size_t)row * N;
    float*       o = O + (size_t)row * N;
    __shared__ float red[NTHREADS];

    float m = -3.4e38f;
    for (int c = tid; c < N; c += NTHREADS) m = fmaxf(m, e[c]);
    red[tid] = m; __syncthreads();
    for (int s = NTHREADS / 2; s > 0; s >>= 1) {
        if (tid < s) red[tid] = fmaxf(red[tid], red[tid + s]);
        __syncthreads();
    }
    m = red[0]; __syncthreads();

    float sum = 0.0f;
    for (int c = tid; c < N; c += NTHREADS) sum += __expf(e[c] - m);
    red[tid] = sum; __syncthreads();
    for (int s = NTHREADS / 2; s > 0; s >>= 1) {
        if (tid < s) red[tid] += red[tid + s];
        __syncthreads();
    }
    const float inv = 1.0f / red[0];

    for (int c = tid; c < N; c += NTHREADS) o[c] = __expf(e[c] - m) * inv;
}

// ---------------------------------------------------------------------------
// Column softmax in place: E[:,n] = softmax over rows. One block per 32 cols;
// lanes map to consecutive columns so every global access is coalesced.
// Online (max,sum) accumulation -> 1 read pass + 1 read/write pass.
// ---------------------------------------------------------------------------
__global__ __launch_bounds__(NTHREADS)
void col_softmax_inplace(float* __restrict__ E, int M, int N)
{
    const int tid  = threadIdx.x;
    const int cl   = tid & 31;               // column within block's 32
    const int grp  = tid >> 5;               // row group 0..7
    const int col  = blockIdx.x * 32 + cl;

    float m = -3.4e38f, s = 0.0f;
    for (int r = grp; r < M; r += 8) {
        const float v  = E[(size_t)r * N + col];
        const float mn = fmaxf(m, v);
        s = s * __expf(m - mn) + __expf(v - mn);
        m = mn;
    }
    __shared__ float sm[8][32];
    __shared__ float ssum[8][32];
    sm[grp][cl] = m; ssum[grp][cl] = s;
    __syncthreads();
    if (grp == 0) {
        float M0 = sm[0][cl], S0 = ssum[0][cl];
#pragma unroll
        for (int g = 1; g < 8; ++g) {
            const float mg = sm[g][cl], sg = ssum[g][cl];
            const float mn = fmaxf(M0, mg);
            S0 = S0 * __expf(M0 - mn) + sg * __expf(mg - mn);
            M0 = mn;
        }
        sm[0][cl] = M0; ssum[0][cl] = 1.0f / S0;
    }
    __syncthreads();
    const float Mf  = sm[0][cl];
    const float inv = ssum[0][cl];
    for (int r = grp; r < M; r += 8) {
        const size_t idx = (size_t)r * N + col;
        E[idx] = __expf(E[idx] - Mf) * inv;
    }
}

// ---------------------------------------------------------------------------
// Orchestration
// ---------------------------------------------------------------------------
extern "C" void kernel_launch(void* const* d_in, const int* in_sizes, int n_in,
                              void* d_out, int out_size, void* d_ws, size_t ws_size,
                              hipStream_t stream)
{
    const float* inp = (const float*)d_in[0];
    const float* vis = (const float*)d_in[1];
    const float* w1w = (const float*)d_in[2];
    const float* b1w = (const float*)d_in[3];
    const float* w2w = (const float*)d_in[4];
    const float* w1v = (const float*)d_in[5];
    const float* b1v = (const float*)d_in[6];
    const float* w2v = (const float*)d_in[7];
    const float* wq  = (const float*)d_in[8];
    const float* bq  = (const float*)d_in[9];
    const float* wk  = (const float*)d_in[10];
    const float* bk  = (const float*)d_in[11];
    const float* wvv = (const float*)d_in[12];
    const float* bvv = (const float*)d_in[13];
    const float* wvw = (const float*)d_in[14];
    const float* bvw = (const float*)d_in[15];

    // Workspace layout (floats). Peak = 76M floats = 304 MB.
    float* ws   = (float*)d_ws;
    float* h    = ws;                               // [NM,SS] (reused by both GCNs)
    float* word = ws + (size_t) 2 * 1024 * 1024;    // [NM,SS]
    float* newv = ws + (size_t) 4 * 1024 * 1024;    // [NM,SS]
    float* Q    = ws + (size_t) 6 * 1024 * 1024;    // [NM,HH]
    float* Km   = ws + (size_t) 7 * 1024 * 1024;    // [NM,HH]
    float* pvv  = ws + (size_t) 8 * 1024 * 1024;    // [NM,SS]
    float* pvw  = ws + (size_t)10 * 1024 * 1024;    // [NM,SS]
    float* E    = ws + (size_t)12 * 1024 * 1024;    // [NM,NM] energy/attention

    float* out0 = (float*)d_out;                    // class_node [NM,SS]
    float* out1 = out0 + (size_t)NM * SS;           // new_v      [NM,SS]

    const dim3 blk(NTHREADS);
    const dim3 gS (SS / BN, NM / BM);               // 2 x 64
    const dim3 gH (HH / BN, NM / BM);               // 1 x 64
    const dim3 gE (NM / BN, NM / BM);               // 64 x 64

    // word GCN: h = relu(W1w*inp + b1w[row] + inp); word = h * w2wT
    gemm_wmma<false,false><<<gS, blk, 0, stream>>>(w1w, NM, inp, SS, h, SS, b1w, 1, inp, 1, NM, SS, NM);
    gemm_wmma<false,true ><<<gS, blk, 0, stream>>>(h, SS, w2w, SS, word, SS, nullptr, 0, nullptr, 0, NM, SS, SS);
    // vis GCN
    gemm_wmma<false,false><<<gS, blk, 0, stream>>>(w1v, NM, vis, SS, h, SS, b1v, 1, vis, 1, NM, SS, NM);
    gemm_wmma<false,true ><<<gS, blk, 0, stream>>>(h, SS, w2v, SS, newv, SS, nullptr, 0, nullptr, 0, NM, SS, SS);
    // Q, K
    gemm_wmma<false,true ><<<gH, blk, 0, stream>>>(word, SS, wq, SS, Q,  HH, bq, 2, nullptr, 0, NM, HH, SS);
    gemm_wmma<false,true ><<<gH, blk, 0, stream>>>(newv, SS, wk, SS, Km, HH, bk, 2, nullptr, 0, NM, HH, SS);
    // pv_vis, pv_word
    gemm_wmma<false,true ><<<gS, blk, 0, stream>>>(newv, SS, wvv, SS, pvv, SS, bvv, 2, nullptr, 0, NM, SS, SS);
    gemm_wmma<false,true ><<<gS, blk, 0, stream>>>(word, SS, wvw, SS, pvw, SS, bvw, 2, nullptr, 0, NM, SS, SS);
    // energy E = Q * K^T, row-softmax in place -> att_vis
    gemm_wmma<false,true ><<<gE, blk, 0, stream>>>(Q, HH, Km, HH, E, NM, nullptr, 0, nullptr, 0, NM, NM, HH);
    row_softmax<<<dim3(NM), blk, 0, stream>>>(E, E, NM);
    // class_node = att_vis * pv_vis + word
    gemm_wmma<false,false><<<gS, blk, 0, stream>>>(E, NM, pvv, SS, out0, SS, nullptr, 0, word, 0, NM, SS, NM);
    // recompute E (17 GFLOP, cheaper than keeping a second 256 MB buffer),
    // col-softmax in place -> att_word
    gemm_wmma<false,true ><<<gE, blk, 0, stream>>>(Q, HH, Km, HH, E, NM, nullptr, 0, nullptr, 0, NM, NM, HH);
    col_softmax_inplace<<<dim3(NM / 32), blk, 0, stream>>>(E, NM, NM);
    // new_v_out = att_word^T * pv_word + newv   (A-transpose path)
    gemm_wmma<true ,false><<<gS, blk, 0, stream>>>(E, NM, pvw, SS, out1, SS, nullptr, 0, newv, 0, NM, SS, NM);
}
